// MHABlock_82678120448109
// MI455X (gfx1250) — compile-verified, matrix operand order checked by reading, not run on hardware
//
#include <hip/hip_runtime.h>
#include <hip/hip_bf16.h>

typedef __attribute__((ext_vector_type(16))) _Float16 v16h;
typedef __attribute__((ext_vector_type(8)))  _Float16 h8;
typedef __attribute__((ext_vector_type(8)))  float    v8f;

#define B_  2
#define S_  2048
#define D_  1536
#define H_  8
#define CQK 128
#define CV  192
#define BS  (B_ * S_)      // 4096
#define NQ  (H_ * CQK)     // 1024
#define NO  (H_ * CV)      // 1536

// ---------------------------------------------------------------------------
// WMMA A-fragment loader (also used for B via B^T, identical lane layout).
// 16-bit A 16x32 layout: lanes 0-15 row M=lane, VGPR0-3 = K 0..7, VGPR4-7 =
// K 16..23 ; lanes 16-31 same rows, K 8..15 / 24..31.
// ---------------------------------------------------------------------------
__device__ __forceinline__ v16h frag_a(const _Float16* base, int stride) {
  const int lane = threadIdx.x & 31;
  const int r = lane & 15;
  const int hi = lane >> 4;
  const h8 lo = *(const h8*)(base + (size_t)r * stride + hi * 8);
  const h8 up = *(const h8*)(base + (size_t)r * stride + 16 + hi * 8);
  v16h a;
#pragma unroll
  for (int j = 0; j < 8; ++j) { a[j] = lo[j]; a[j + 8] = up[j]; }
  return a;
}

__device__ __forceinline__ v8f wmma_f16(v16h a, v16h b, v8f c) {
  return __builtin_amdgcn_wmma_f32_16x16x32_f16(false, a, false, b,
                                                (short)0, c, false, false);
}

// ---------------------------------------------------------------------------
// 1) RMS-norm of input, fp32 -> f16
// ---------------------------------------------------------------------------
__global__ __launch_bounds__(256)
void rmsnorm_in_kernel(const float* __restrict__ x, const float* __restrict__ g,
                       _Float16* __restrict__ h) {
  const int row = blockIdx.x;
  const int t = threadIdx.x;
  const float* xr = x + (size_t)row * D_;
  float vals[6];
  float ss = 0.f;
#pragma unroll
  for (int j = 0; j < 6; ++j) { vals[j] = xr[t + j * 256]; ss += vals[j] * vals[j]; }
  __shared__ float red[256];
  red[t] = ss;
  __syncthreads();
  for (int st = 128; st > 0; st >>= 1) {
    if (t < st) red[t] += red[t + st];
    __syncthreads();
  }
  const float inv = rsqrtf(red[0] * (1.0f / D_) + 1e-6f);
  _Float16* hr = h + (size_t)row * D_;
#pragma unroll
  for (int j = 0; j < 6; ++j)
    hr[t + j * 256] = (_Float16)(vals[j] * inv * g[t + j * 256]);
}

// ---------------------------------------------------------------------------
// 2) Weight transpose + convert: W[K,N] fp32 -> Wt[N,K] f16
// ---------------------------------------------------------------------------
__global__ void transpose_cvt_kernel(const float* __restrict__ W,
                                     _Float16* __restrict__ Wt, int K, int N) {
  size_t i = (size_t)blockIdx.x * blockDim.x + threadIdx.x;
  size_t total = (size_t)K * N;
  if (i >= total) return;
  size_t n = i % N, k = i / N;
  Wt[n * K + k] = (_Float16)W[i];
}

// ---------------------------------------------------------------------------
// 3) Generic WMMA GEMM: C[M,N] = A[M,K] * Bt[N,K]^T  (fp32 accumulate)
//    block = 4 waves, each wave owns a 16x64 tile; Bt tile staged in LDS and
//    reused by all waves. K must be a multiple of 128.
// ---------------------------------------------------------------------------
__global__ __launch_bounds__(128)
void gemm_kernel(const _Float16* __restrict__ A, const _Float16* __restrict__ Bt,
                 float* __restrict__ C, int M, int N, int K) {
  __shared__ __align__(16) _Float16 Bl[64][128];
  const int tid = threadIdx.x;
  const int lane = tid & 31, wv = tid >> 5;
  const int r = lane & 15, hi = lane >> 4;
  const int n0 = blockIdx.x * 64;
  const int m0 = blockIdx.y * 64 + wv * 16;

  v8f cfr[4] = {};
  for (int k0 = 0; k0 < K; k0 += 128) {
    __syncthreads();
    for (int i = tid; i < 64 * 16; i += 128) {
      int row = i >> 4, c8 = (i & 15) * 8;
      *(h8*)&Bl[row][c8] = *(const h8*)(Bt + (size_t)(n0 + row) * K + k0 + c8);
    }
    __syncthreads();
#pragma unroll
    for (int kk = 0; kk < 4; ++kk) {
      v16h a = frag_a(A + (size_t)m0 * K + k0 + kk * 32, K);
#pragma unroll
      for (int nt = 0; nt < 4; ++nt) {
        v16h b = frag_a(&Bl[nt * 16][kk * 32], 128);
        cfr[nt] = wmma_f16(a, b, cfr[nt]);
      }
    }
  }
#pragma unroll
  for (int nt = 0; nt < 4; ++nt)
#pragma unroll
    for (int v = 0; v < 8; ++v)
      C[(size_t)(m0 + v + 8 * hi) * N + n0 + nt * 16 + r] = cfr[nt][v];
}

// ---------------------------------------------------------------------------
// 4) LayerNorm(128) + RoPE for Q/K.  grid = (B*S, heads), block = 128.
//    Q out: [B,H,S,128] f16 ; K out: [B,S,128] f16
// ---------------------------------------------------------------------------
__global__ __launch_bounds__(128)
void qk_norm_rope_kernel(const float* __restrict__ pre, int ld,
                         const float* __restrict__ sc, const float* __restrict__ of,
                         _Float16* __restrict__ out, int is_q) {
  const int bs = blockIdx.x, h = blockIdx.y, c = threadIdx.x;
  const int b = bs >> 11, s = bs & (S_ - 1);
  const float x = pre[(size_t)bs * ld + h * CQK + c];
  __shared__ float vals[128];
  __shared__ float red[128];
  red[c] = x;
  __syncthreads();
  for (int st = 64; st > 0; st >>= 1) { if (c < st) red[c] += red[c + st]; __syncthreads(); }
  const float mean = red[0] * (1.f / 128.f);
  __syncthreads();
  const float d = x - mean;
  red[c] = d * d;
  __syncthreads();
  for (int st = 64; st > 0; st >>= 1) { if (c < st) red[c] += red[c + st]; __syncthreads(); }
  const float var = red[0] * (1.f / 128.f);
  const float nv = d * rsqrtf(var + 1e-5f) * sc[c] + of[c];
  vals[c] = nv;
  __syncthreads();
  // RoPE: inv_freq[f] = 1 / (f + 8129^(f/63)),  f = c>>1, theta = s*inv_freq
  const int f = c >> 1;
  const float inv_fr = 1.0f / ((float)f + exp2f((float)f * (12.988859f / 63.0f)));
  const float th = (float)s * inv_fr;
  const float cs = cosf(th), sn = sinf(th);
  const float rot = (c & 1) ? vals[c - 1] : -vals[c + 1];
  const float res = nv * cs + rot * sn;
  const size_t oi = is_q ? (((size_t)(b * H_ + h) * S_ + s) * CQK + c)
                         : ((size_t)bs * CQK + c);
  out[oi] = (_Float16)res;
}

// ---------------------------------------------------------------------------
// 5) LayerNorm(192) for V, written transposed: Vt[B,192,S] f16
// ---------------------------------------------------------------------------
__global__ __launch_bounds__(192)
void v_norm_kernel(const float* __restrict__ pre, const float* __restrict__ sc,
                   const float* __restrict__ of, _Float16* __restrict__ vt) {
  const int bs = blockIdx.x, c = threadIdx.x;
  const int b = bs >> 11, s = bs & (S_ - 1);
  const float x = pre[(size_t)bs * CV + c];
  __shared__ float red[192];
  red[c] = x;
  __syncthreads();
  for (int st = 128; st > 0; st >>= 1) {
    if (c < st && c + st < 192) red[c] += red[c + st];
    __syncthreads();
  }
  const float mean = red[0] * (1.f / 192.f);
  __syncthreads();
  const float d = x - mean;
  red[c] = d * d;
  __syncthreads();
  for (int st = 128; st > 0; st >>= 1) {
    if (c < st && c + st < 192) red[c] += red[c + st];
    __syncthreads();
  }
  const float var = red[0] * (1.f / 192.f);
  vt[((size_t)b * CV + c) * S_ + s] =
      (_Float16)(d * rsqrtf(var + 1e-5f) * sc[c] + of[c]);
}

// ---------------------------------------------------------------------------
// 6) Flash attention with bias + tanh softcap.  grid (S/64, H, B), 4 waves.
//    Each wave: 16 query rows, online softmax, PV into 12 fp32 C-fragments.
// ---------------------------------------------------------------------------
__global__ __launch_bounds__(128)
void attn_kernel(const _Float16* __restrict__ q16,   // [B,H,S,128]
                 const _Float16* __restrict__ k16,   // [B,S,128]
                 const _Float16* __restrict__ vt16,  // [B,192,S]
                 const float* __restrict__ bias,     // [B,H,S,S]
                 _Float16* __restrict__ y16)         // [B*S, 1536]
{
  __shared__ __align__(16) _Float16 Kl[64][128];
  __shared__ __align__(16) _Float16 Vl[192][80];
  __shared__ __align__(16) _Float16 Pl[4][16][80];

  const int tid = threadIdx.x;
  const int lane = tid & 31, wv = tid >> 5;
  const int r = lane & 15, hi = lane >> 4;
  const int b = blockIdx.z, h = blockIdx.y;
  const int q0 = blockIdx.x * 64 + wv * 16;

  const _Float16* qp = q16 + (((size_t)(b * H_ + h) * S_) + q0) * CQK;
  v16h qa[4];
#pragma unroll
  for (int kk = 0; kk < 4; ++kk) qa[kk] = frag_a(qp + kk * 32, CQK);

  float rm[8], rl[8];
#pragma unroll
  for (int v = 0; v < 8; ++v) { rm[v] = -1e30f; rl[v] = 0.f; }
  v8f acc[12] = {};

  const _Float16* kp = k16 + (size_t)b * S_ * CQK;
  const _Float16* vp = vt16 + (size_t)b * CV * S_;
  const float* bp = bias + (((size_t)(b * H_ + h)) * S_ + q0) * S_;

  for (int kb = 0; kb < S_ / 64; ++kb) {
    __syncthreads();
    // stage K block [64 x 128]
    for (int i = tid; i < 64 * 16; i += 128) {
      int row = i >> 4, c8 = (i & 15) * 8;
      *(h8*)&Kl[row][c8] = *(const h8*)(kp + (size_t)(kb * 64 + row) * CQK + c8);
    }
    // stage V^T block [192 x 64]
    for (int i = tid; i < 192 * 8; i += 128) {
      int row = i >> 3, c8 = (i & 7) * 8;
      *(h8*)&Vl[row][c8] = *(const h8*)(vp + (size_t)row * S_ + kb * 64 + c8);
    }
    if (kb + 1 < S_ / 64)  // hint next K block into cache
      __builtin_prefetch(kp + (size_t)(kb + 1) * 64 * CQK + tid * 64, 0, 0);
    __syncthreads();

    // S = Q K^T  (4 tiles of 16 cols), then bias + softcap
    v8f sc[4];
#pragma unroll
    for (int nt = 0; nt < 4; ++nt) {
      v8f c = {};
#pragma unroll
      for (int kk = 0; kk < 4; ++kk) {
        v16h bf = frag_a(&Kl[nt * 16][kk * 32], 128);
        c = wmma_f16(qa[kk], bf, c);
      }
#pragma unroll
      for (int v = 0; v < 8; ++v) {
        const int m = v + 8 * hi;
        float x = c[v] * 0.08838834764831845f +
                  bp[(size_t)m * S_ + kb * 64 + nt * 16 + r];
        c[v] = tanhf(x * 0.2f) * 5.0f;
      }
      sc[nt] = c;
    }

    // online softmax (rows live across the 16-lane half: reduce via shfl_xor)
#pragma unroll
    for (int v = 0; v < 8; ++v) {
      float mx = fmaxf(fmaxf(sc[0][v], sc[1][v]), fmaxf(sc[2][v], sc[3][v]));
#pragma unroll
      for (int o = 8; o > 0; o >>= 1) mx = fmaxf(mx, __shfl_xor(mx, o, 32));
      const float nm = fmaxf(rm[v], mx);
      const float corr = __expf(rm[v] - nm);
      rm[v] = nm;
      float ssum = 0.f;
#pragma unroll
      for (int nt = 0; nt < 4; ++nt) {
        const float e = __expf(sc[nt][v] - nm);
        sc[nt][v] = e;
        ssum += e;
      }
#pragma unroll
      for (int o = 8; o > 0; o >>= 1) ssum += __shfl_xor(ssum, o, 32);
      rl[v] = rl[v] * corr + ssum;
#pragma unroll
      for (int t = 0; t < 12; ++t) acc[t][v] *= corr;
    }

    // reshape P (C-layout) -> A-layout via per-wave LDS tile
#pragma unroll
    for (int nt = 0; nt < 4; ++nt)
#pragma unroll
      for (int v = 0; v < 8; ++v)
        Pl[wv][v + 8 * hi][nt * 16 + r] = (_Float16)sc[nt][v];

    const v16h pa0 = frag_a(&Pl[wv][0][0], 80);
    const v16h pa1 = frag_a(&Pl[wv][0][32], 80);
#pragma unroll
    for (int t = 0; t < 12; ++t) {
      v16h b0 = frag_a(&Vl[t * 16][0], 80);
      acc[t] = wmma_f16(pa0, b0, acc[t]);
      v16h b1 = frag_a(&Vl[t * 16][32], 80);
      acc[t] = wmma_f16(pa1, b1, acc[t]);
    }
  }

  // normalize and store y f16: [B*S, H*CV]
  _Float16* yp = y16 + ((size_t)b * S_ + q0) * NO + h * CV;
#pragma unroll
  for (int v = 0; v < 8; ++v) {
    const float inv = 1.0f / rl[v];
    const int m = v + 8 * hi;
#pragma unroll
    for (int t = 0; t < 12; ++t)
      yp[(size_t)m * NO + t * 16 + r] = (_Float16)(acc[t][v] * inv);
  }
}

// ---------------------------------------------------------------------------
// 7) bias add + output RMS-norm, fp32 out
// ---------------------------------------------------------------------------
__global__ __launch_bounds__(256)
void rmsnorm_out_kernel(const float* __restrict__ o, const float* __restrict__ bo,
                        const float* __restrict__ g, float* __restrict__ out) {
  const int row = blockIdx.x;
  const int t = threadIdx.x;
  const float* orow = o + (size_t)row * D_;
  float vals[6];
  float ss = 0.f;
#pragma unroll
  for (int j = 0; j < 6; ++j) {
    vals[j] = orow[t + j * 256] + bo[t + j * 256];
    ss += vals[j] * vals[j];
  }
  __shared__ float red[256];
  red[t] = ss;
  __syncthreads();
  for (int st = 128; st > 0; st >>= 1) {
    if (t < st) red[t] += red[t + st];
    __syncthreads();
  }
  const float inv = rsqrtf(red[0] * (1.0f / D_) + 1e-6f);
  float* outr = out + (size_t)row * D_;
#pragma unroll
  for (int j = 0; j < 6; ++j)
    outr[t + j * 256] = vals[j] * inv * g[t + j * 256];
}

// ---------------------------------------------------------------------------
extern "C" void kernel_launch(void* const* d_in, const int* in_sizes, int n_in,
                              void* d_out, int out_size, void* d_ws, size_t ws_size,
                              hipStream_t stream) {
  const float* x       = (const float*)d_in[0];
  const float* abias   = (const float*)d_in[1];
  const float* rms_in  = (const float*)d_in[2];
  const float* wq      = (const float*)d_in[3];
  const float* wk      = (const float*)d_in[4];
  const float* wv      = (const float*)d_in[5];
  const float* q_sc    = (const float*)d_in[6];
  const float* q_of    = (const float*)d_in[7];
  const float* k_sc    = (const float*)d_in[8];
  const float* k_of    = (const float*)d_in[9];
  const float* v_sc    = (const float*)d_in[10];
  const float* v_of    = (const float*)d_in[11];
  const float* wo      = (const float*)d_in[12];
  const float* bo      = (const float*)d_in[13];
  const float* rms_out = (const float*)d_in[14];
  float* out = (float*)d_out;

  char* w = (char*)d_ws;
  size_t off = 0;
  auto take = [&](size_t bytes) -> void* {
    void* p = w + off;
    off += (bytes + 255) & ~(size_t)255;
    return p;
  };
  _Float16* h16  = (_Float16*)take((size_t)BS * D_ * 2);
  _Float16* wqt  = (_Float16*)take((size_t)NQ * D_ * 2);
  _Float16* wkt  = (_Float16*)take((size_t)CQK * D_ * 2);
  _Float16* wvt  = (_Float16*)take((size_t)CV * D_ * 2);
  _Float16* wot  = (_Float16*)take((size_t)D_ * NO * 2);
  float*    qpre = (float*)take((size_t)BS * NQ * 4);
  float*    kpre = (float*)take((size_t)BS * CQK * 4);
  float*    vpre = (float*)take((size_t)BS * CV * 4);
  _Float16* q16  = (_Float16*)take((size_t)B_ * H_ * S_ * CQK * 2);
  _Float16* k16  = (_Float16*)take((size_t)B_ * S_ * CQK * 2);
  _Float16* vt16 = (_Float16*)take((size_t)B_ * CV * S_ * 2);
  _Float16* y16  = (_Float16*)take((size_t)BS * NO * 2);
  float*    opre = (float*)take((size_t)BS * D_ * 4);

  // 1) input RMS norm -> f16
  rmsnorm_in_kernel<<<BS, 256, 0, stream>>>(x, rms_in, h16);

  // 2) weight transpose+convert
  {
    size_t t;
    t = (size_t)D_ * NQ;
    transpose_cvt_kernel<<<(t + 255) / 256, 256, 0, stream>>>(wq, wqt, D_, NQ);
    t = (size_t)D_ * CQK;
    transpose_cvt_kernel<<<(t + 255) / 256, 256, 0, stream>>>(wk, wkt, D_, CQK);
    t = (size_t)D_ * CV;
    transpose_cvt_kernel<<<(t + 255) / 256, 256, 0, stream>>>(wv, wvt, D_, CV);
    t = (size_t)NO * D_;
    transpose_cvt_kernel<<<(t + 255) / 256, 256, 0, stream>>>(wo, wot, NO, D_);
  }

  // 3) QKV projections (WMMA GEMM)
  gemm_kernel<<<dim3(NQ / 64, BS / 64), 128, 0, stream>>>(h16, wqt, qpre, BS, NQ, D_);
  gemm_kernel<<<dim3(CQK / 64, BS / 64), 128, 0, stream>>>(h16, wkt, kpre, BS, CQK, D_);
  gemm_kernel<<<dim3(CV / 64, BS / 64), 128, 0, stream>>>(h16, wvt, vpre, BS, CV, D_);

  // 4) LN + RoPE for q, k ; LN for v (v stored transposed)
  qk_norm_rope_kernel<<<dim3(BS, H_), 128, 0, stream>>>(qpre, NQ, q_sc, q_of, q16, 1);
  qk_norm_rope_kernel<<<dim3(BS, 1), 128, 0, stream>>>(kpre, CQK, k_sc, k_of, k16, 0);
  v_norm_kernel<<<BS, 192, 0, stream>>>(vpre, v_sc, v_of, vt16);

  // 5) flash attention with bias + softcap (WMMA)
  attn_kernel<<<dim3(S_ / 64, H_, B_), 128, 0, stream>>>(q16, k16, vt16, abias, y16);

  // 6) output projection (WMMA GEMM) + bias + output RMS norm
  gemm_kernel<<<dim3(D_ / 64, BS / 64), 128, 0, stream>>>(y16, wot, opre, BS, D_, NO);
  rmsnorm_out_kernel<<<BS, 256, 0, stream>>>(opre, bo, rms_out, out);
}
